// DVAE_68247030333747
// MI455X (gfx1250) — matrix operand order repaired and checked by reading, not compiled, not run on hardware
//
#include <hip/hip_runtime.h>
#include <math.h>

// ---------------------------------------------------------------------------
// Types for CDNA5 WMMA (wave32, 16x16x32 bf16 -> f32)
// ---------------------------------------------------------------------------
typedef __attribute__((ext_vector_type(16))) __bf16 v16bf;
typedef __attribute__((ext_vector_type(8)))  float  v8f;

// ---------------------------------------------------------------------------
// bf16 helpers (bit tricks; bf16 <-> f32)
// ---------------------------------------------------------------------------
__device__ __forceinline__ float bf16_to_f(unsigned short b) {
    return __uint_as_float(((unsigned int)b) << 16);
}
__device__ __forceinline__ unsigned short f_to_bf16(float f) {
    unsigned int u = __float_as_uint(f);
    u += 0x7FFFu + ((u >> 16) & 1u);   // round-to-nearest-even
    return (unsigned short)(u >> 16);
}

// ===========================================================================
// 1) Encoder convs (3x3, stride 8 / 16, Cin=3, Cout=4) -- tiny, direct.
// ===========================================================================
__global__ void k_conv_enc(const float* __restrict__ x, const float* __restrict__ w,
                           const float* __restrict__ b, float* __restrict__ out,
                           int stride, int Ho, int Wo_) {
    int idx = blockIdx.x * 256 + threadIdx.x;
    int total = 8 * 4 * Ho * Wo_;
    if (idx >= total) return;
    int xo = idx % Wo_; int t = idx / Wo_;
    int yo = t % Ho;    t /= Ho;
    int co = t & 3;     int bimg = t >> 2;
    const float* I = x + (size_t)bimg * 3 * 65536;
    float s = b[co];
    for (int ci = 0; ci < 3; ++ci)
        for (int ky = 0; ky < 3; ++ky) {
            int sy = yo * stride + ky - 1; if (sy < 0 || sy > 255) continue;
            for (int kx = 0; kx < 3; ++kx) {
                int sx = xo * stride + kx - 1; if (sx < 0 || sx > 255) continue;
                s += w[((co * 3 + ci) * 3 + ky) * 3 + kx] * I[(ci * 256 + sy) * 256 + sx];
            }
        }
    out[idx] = s;
}

// ===========================================================================
// 2) Per-patch KDE entropy. One block (256 threads) per 16x16 patch.
// ===========================================================================
__global__ void __launch_bounds__(256) k_entropy(const float* __restrict__ x,
                                                 float* __restrict__ ent_ws,
                                                 float* __restrict__ ent_out) {
    __shared__ float sg[256];
    __shared__ float red[256];
    int t = threadIdx.x;
    int patch = blockIdx.x;                 // b*256 + pi*16 + pj
    int pj = patch & 15, pi = (patch >> 4) & 15, bimg = patch >> 8;
    int yy = pi * 16 + (t >> 4), xx = pj * 16 + (t & 15);
    const float* I = x + (size_t)bimg * 3 * 65536;
    int pix = yy * 256 + xx;
    sg[t] = 0.299f * I[pix] + 0.587f * I[65536 + pix] + 0.114f * I[131072 + pix];
    __syncthreads();
    float bin = (float)t * (1.0f / 255.0f);
    float s = 0.0f;
    for (int p = 0; p < 256; ++p) {
        float d = (sg[p] - bin) * 100.0f;   // / sigma (0.01)
        s += __expf(-0.5f * d * d);
    }
    float pdf = s * (1.0f / 256.0f);
    red[t] = pdf; __syncthreads();
    for (int o = 128; o > 0; o >>= 1) { if (t < o) red[t] += red[t + o]; __syncthreads(); }
    float tot = red[0]; __syncthreads();
    float pn = fmaxf(pdf / tot, 1e-10f);
    red[t] = -pn * __log2f(pn); __syncthreads();
    for (int o = 128; o > 0; o >>= 1) { if (t < o) red[t] += red[t + o]; __syncthreads(); }
    if (t == 0) { ent_ws[patch] = red[0]; ent_out[patch] = red[0]; }
}

// ===========================================================================
// 3) Median of 2048 entropies via bitonic sort in LDS (1 block, 1024 thr).
// ===========================================================================
__global__ void __launch_bounds__(1024) k_median(const float* __restrict__ ent,
                                                 float* __restrict__ thr) {
    __shared__ float a[2048];
    int t = threadIdx.x;
    a[t] = ent[t]; a[t + 1024] = ent[t + 1024];
    __syncthreads();
    for (int k = 2; k <= 2048; k <<= 1) {
        for (int j = k >> 1; j > 0; j >>= 1) {
            for (int base = 0; base < 2048; base += 1024) {
                int i = base + t;
                int ixj = i ^ j;
                if (ixj > i) {
                    bool up = ((i & k) == 0);
                    float ai = a[i], aj = a[ixj];
                    if ((ai > aj) == up) { a[i] = aj; a[ixj] = ai; }
                }
            }
            __syncthreads();
        }
    }
    if (t == 0) thr[0] = 0.5f * (a[1023] + a[1024]);
}

// ===========================================================================
// 4) Routing: routed = g*fine + (1-g)*up(coarse); also emits grain map.
// ===========================================================================
__global__ void k_route(const float* __restrict__ fine, const float* __restrict__ coarse,
                        const float* __restrict__ ent, const float* __restrict__ thr,
                        float* __restrict__ routed_out, float* __restrict__ routed_ws,
                        float* __restrict__ grain_out) {
    int idx = blockIdx.x * 256 + threadIdx.x;   // [8][4][32][32]
    if (idx >= 32768) return;
    int x = idx & 31, y = (idx >> 5) & 31, c = (idx >> 10) & 3, b = idx >> 12;
    float T = thr[0];
    float e = ent[(b * 16 + (y >> 1)) * 16 + (x >> 1)];
    float g = (e > T) ? 1.0f : 0.0f;
    float r = g * fine[idx] + (1.0f - g) * coarse[((b * 4 + c) * 16 + (y >> 1)) * 16 + (x >> 1)];
    routed_out[idx] = r;
    routed_ws[idx]  = r;
    if (c == 0 && (y & 1) == 0 && (x & 1) == 0)
        grain_out[(b * 16 + (y >> 1)) * 16 + (x >> 1)] = g;
}

// ===========================================================================
// 5) Weight prepack into CDNA5 bf16 16x32 A-fragment layout (ISA 7.12.2),
//    K ordered as (tap, ci). Handles ConvTranspose flip/transpose.
//    Fragment fg = (co_tile*9 + tap)*cslabs + cs ; elem = fg*512 + lane*16 + e
// ===========================================================================
__global__ void k_prepack(const float* __restrict__ w, unsigned short* __restrict__ out,
                          int Cout, int Cin, int deconv) {
    int idx = blockIdx.x * 256 + threadIdx.x;
    int total = Cout * Cin * 9;
    if (idx >= total) return;
    int e    = idx & 15;
    int lane = (idx >> 4) & 31;
    int fg   = idx >> 9;
    int cslabs = Cin >> 5;
    int cs  = fg % cslabs;
    int tap = (fg / cslabs) % 9;
    int cot = fg / (cslabs * 9);
    int m = lane & 15, half = lane >> 4;
    int j = e >> 1, bit = e & 1;
    // A-matrix 16x32 bf16 layout: K = half*16 + 2j+bit (contiguous, matches B)
    int l = half * 16 + 2 * j + bit;
    int ci = cs * 32 + l;
    int ky = tap / 3, kx = tap % 3;
    int co = cot * 16 + m;
    float v;
    if (deconv)  // wt[co,ci,ky,kx] = w[ci,co,2-ky,2-kx]
        v = w[(((size_t)ci * Cout + co) * 3 + (2 - ky)) * 3 + (2 - kx)];
    else
        v = w[(((size_t)co * Cin + ci) * 3 + ky) * 3 + kx];
    out[idx] = f_to_bf16(v);
}

// ===========================================================================
// 6) Zero-stuff (materialize lhs_dilation=2) on pair-interleaved bf16 dwords.
// ===========================================================================
__global__ void k_zstuff(const unsigned int* __restrict__ in, unsigned int* __restrict__ out,
                         int H, int W, size_t total) {
    size_t idx = (size_t)blockIdx.x * 256 + threadIdx.x;
    if (idx >= total) return;
    int W2 = 2 * W, H2 = 2 * H;
    size_t ox = idx % W2; size_t t = idx / W2;
    size_t oy = t % H2;   t /= H2;            // t = b*Cp + cp
    unsigned int v = 0;
    if (((ox & 1) == 0) && ((oy & 1) == 0))
        v = in[(t * H + (oy >> 1)) * W + (ox >> 1)];
    out[idx] = v;
}

// ===========================================================================
// 7) Main implicit-GEMM conv3x3 pad1 stride1, bf16 WMMA 16x16x32.
//    Templated on CSLABS = Cin/32 and LOGW (square power-of-two spatial).
//    Each wave computes TWO adjacent Cout tiles (32 x 16 output): B fragment
//    is built once and fed to two back-to-back WMMAs.
//    LOGW >= 7: each 32-channel K-slab (16 ch-pairs x 3 halo rows x 130 cols,
//    ~25 KB) is staged into LDS once per block with the CDNA5 async DMA path
//    (GLOBAL_LOAD_ASYNC_TO_LDS_B32, ASYNCcnt; ISA 08 §4) -- no VGPR round
//    trip -- then all 9 tap fragments are built from LDS, cutting global-side
//    activation reads ~3x (HBM-critical on the 256^2 stages whose footprint
//    exceeds L2). Consume point: s_wait_asynccnt 0 + workgroup barrier.
//    LOGW == 6: direct global gather (activations are L2-resident anyway).
//    Grid: (H*W/128, Cout/32, 8 images).
// ===========================================================================
template <int CSLABS, int LOGW>
__global__ void __launch_bounds__(256) k_conv_wmma(
    const unsigned short* __restrict__ apack,
    const unsigned int*  __restrict__ act,     // [B][Cin/2][H][W] dwords
    const float*         __restrict__ bias,
    unsigned short*      __restrict__ out,     // [B][Cout/2][H][W][2] bf16
    int Cout, int relu)
{
    constexpr int W  = 1 << LOGW;
    constexpr int H  = W;
    constexpr int HW = H * W;
    constexpr int Cin = CSLABS * 32;

    const int lane = threadIdx.x & 31;
    const int wave = threadIdx.x >> 5;
    const int nl = lane & 15, half = lane >> 4;
    const int p  = blockIdx.x * 128 + wave * 16 + nl;   // pixel (N index)
    const int py = p >> LOGW, px = p & (W - 1);
    const int cot = blockIdx.y * 2;                     // two adjacent 16-row tiles
    const int img = blockIdx.z;

    const unsigned int* A   = act + (size_t)img * (Cin >> 1) * HW;
    const v16bf*        af0 = ((const v16bf*)apack) + (size_t)cot * 9 * CSLABS * 32 + lane;
    const v16bf*        af1 = af0 + (size_t)9 * CSLABS * 32;

    v8f c0 = {};
    v8f c1 = {};

    if constexpr (LOGW >= 7) {
        // ---- LDS-staged path: block covers one row segment [x0, x0+127] ----
        __shared__ unsigned int tile[16 * 3 * 130];     // ch-pair x dy x dx
        const int y0 = (blockIdx.x * 128) >> LOGW;
        const int x0 = (blockIdx.x * 128) & (W - 1);
        const int col = wave * 16 + nl;                 // px - x0
        for (int cs = 0; cs < CSLABS; ++cs) {
            // cooperative async stage of this 32-ch slab (+halo); OOB -> 0
            for (int i = threadIdx.x; i < 16 * 3 * 130; i += 256) {
                int dx = i % 130; int tq = i / 130;
                int dy = tq % 3;  int cp = tq / 3;
                int gy = y0 + dy - 1, gx = x0 + dx - 1;
                unsigned lds_b = (unsigned)(size_t)&tile[i];   // LDS byte addr
                if (gy >= 0 && gy < H && gx >= 0 && gx < W) {
                    const unsigned int* gp =
                        A + ((size_t)(cs * 16 + cp) << (2 * LOGW)) + (gy << LOGW) + gx;
                    // async DMA: memory -> LDS, tracked by ASYNCcnt
                    asm volatile("global_load_async_to_lds_b32 %0, %1, off"
                                 :: "v"(lds_b),
                                    "v"((unsigned long long)(size_t)gp)
                                 : "memory");
                } else {
                    tile[i] = 0;
                }
            }
            asm volatile("s_wait_asynccnt 0x0" ::: "memory");
            __syncthreads();
#pragma unroll
            for (int tap = 0; tap < 9; ++tap) {
                const int ky = tap / 3, kx = tap % 3;
                union { unsigned int u[8]; v16bf v; } b;
#pragma unroll
                for (int j = 0; j < 8; ++j)
                    b.u[j] = tile[((half * 8 + j) * 3 + ky) * 130 + col + kx];
                v16bf a0 = af0[(tap * CSLABS + cs) * 32];
                v16bf a1 = af1[(tap * CSLABS + cs) * 32];
                c0 = __builtin_amdgcn_wmma_f32_16x16x32_bf16(
                         false, a0, false, b.v, (short)0, c0, false, false);
                c1 = __builtin_amdgcn_wmma_f32_16x16x32_bf16(
                         false, a1, false, b.v, (short)0, c1, false, false);
            }
            __syncthreads();
        }
    } else {
        // ---- direct-gather path (W = 64) ----
        for (int tap = 0; tap < 9; ++tap) {
            int sy = py + tap / 3 - 1;
            int sx = px + tap % 3 - 1;
            bool valid = (sy >= 0) & (sy < H) & (sx >= 0) & (sx < W);
            int pix = (sy << LOGW) + sx;
            const unsigned int* bp = A + half * 8 * HW + pix;
            if (valid) __builtin_prefetch(bp, 0, 1);    // global_prefetch_b8
#pragma unroll
            for (int cs = 0; cs < CSLABS; ++cs) {
                union { unsigned int u[8]; v16bf v; } b;
#pragma unroll
                for (int j = 0; j < 8; ++j)
                    b.u[j] = valid ? bp[(size_t)(cs * 16 + j) * HW] : 0u;
                v16bf a0 = af0[(tap * CSLABS + cs) * 32];
                v16bf a1 = af1[(tap * CSLABS + cs) * 32];
                c0 = __builtin_amdgcn_wmma_f32_16x16x32_bf16(
                         false, a0, false, b.v, (short)0, c0, false, false);
                c1 = __builtin_amdgcn_wmma_f32_16x16x32_bf16(
                         false, a1, false, b.v, (short)0, c1, false, false);
            }
        }
    }

    // Epilogue: C/D layout -> VGPR r: lanes0-15 M=r, lanes16-31 M=r+8
    unsigned short* O = out + (size_t)img * Cout * HW;
#pragma unroll
    for (int r = 0; r < 8; ++r) {
        int co = cot * 16 + r + half * 8;
        float v = c0[r] + bias[co];
        if (relu) v = fmaxf(v, 0.0f);
        O[((size_t)(co >> 1) * HW + p) * 2 + (co & 1)] = f_to_bf16(v);
        int co2 = co + 16;
        float v2 = c1[r] + bias[co2];
        if (relu) v2 = fmaxf(v2, 0.0f);
        O[((size_t)(co2 >> 1) * HW + p) * 2 + (co2 & 1)] = f_to_bf16(v2);
    }
}

// ===========================================================================
// 8) Wi: conv3x3 Cin=4 -> Cout=256 on 32x32 (too thin in K for WMMA).
// ===========================================================================
__global__ void k_conv_wi(const float* __restrict__ in,    // [8][4][32][32]
                          const float* __restrict__ w,     // [256][4][3][3]
                          const float* __restrict__ b,
                          unsigned short* __restrict__ out) {
    int idx = blockIdx.x * 256 + threadIdx.x;              // 8*256*1024
    if (idx >= 8 * 256 * 1024) return;
    int x = idx & 31, y = (idx >> 5) & 31, co = (idx >> 10) & 255, bimg = idx >> 18;
    const float* I = in + (size_t)bimg * 4 * 1024;
    float s = b[co];
    for (int ci = 0; ci < 4; ++ci)
        for (int ky = 0; ky < 3; ++ky) {
            int sy = y + ky - 1; if (sy < 0 || sy > 31) continue;
            for (int kx = 0; kx < 3; ++kx) {
                int sx = x + kx - 1; if (sx < 0 || sx > 31) continue;
                s += w[((co * 4 + ci) * 3 + ky) * 3 + kx] * I[(ci * 32 + sy) * 32 + sx];
            }
        }
    out[(size_t)bimg * 256 * 1024 + ((size_t)(co >> 1) * 1024 + (y * 32 + x)) * 2 + (co & 1)]
        = f_to_bf16(s);
}

// ===========================================================================
// 9) Wo: conv3x3 Cin=128 -> Cout=3 on 256x256, tanh. Direct, bf16-pair reads.
// ===========================================================================
__global__ void k_conv_wo(const unsigned short* __restrict__ h,  // [8][64][65536][2]
                          const float* __restrict__ w,           // [3][128][3][3]
                          const float* __restrict__ b,
                          float* __restrict__ rec) {             // [8][3][256][256]
    int idx = blockIdx.x * 256 + threadIdx.x;                    // 8*3*65536
    if (idx >= 8 * 3 * 65536) return;
    int x = idx & 255, y = (idx >> 8) & 255;
    int co = (idx >> 16) % 3, bimg = (idx >> 16) / 3;
    const unsigned int* H32 = (const unsigned int*)(h + (size_t)bimg * 128 * 65536);
    float s = b[co];
    for (int ky = 0; ky < 3; ++ky) {
        int sy = y + ky - 1; if (sy < 0 || sy > 255) continue;
        for (int kx = 0; kx < 3; ++kx) {
            int sx = x + kx - 1; if (sx < 0 || sx > 255) continue;
            int pix = sy * 256 + sx;
            const float* wr = w + (size_t)co * 128 * 9 + ky * 3 + kx;
            for (int cp = 0; cp < 64; ++cp) {
                unsigned int d = H32[(size_t)cp * 65536 + pix];
                s += wr[(2 * cp) * 9]     * bf16_to_f((unsigned short)(d & 0xffffu));
                s += wr[(2 * cp + 1) * 9] * bf16_to_f((unsigned short)(d >> 16));
            }
        }
    }
    rec[idx] = tanhf(s);
}

// ===========================================================================
// Host-side orchestration
// ===========================================================================
extern "C" void kernel_launch(void* const* d_in, const int* in_sizes, int n_in,
                              void* d_out, int out_size, void* d_ws, size_t ws_size,
                              hipStream_t stream) {
    (void)in_sizes; (void)n_in; (void)out_size; (void)ws_size;
    const float* x   = (const float*)d_in[0];
    const float* We0 = (const float*)d_in[1];  const float* be0 = (const float*)d_in[2];
    const float* We1 = (const float*)d_in[3];  const float* be1 = (const float*)d_in[4];
    const float* Wi  = (const float*)d_in[5];  const float* bi  = (const float*)d_in[6];
    const float* Wd1 = (const float*)d_in[7];  const float* bd1 = (const float*)d_in[8];
    const float* Wc1 = (const float*)d_in[9];  const float* bc1 = (const float*)d_in[10];
    const float* Wd2 = (const float*)d_in[11]; const float* bd2 = (const float*)d_in[12];
    const float* Wc2 = (const float*)d_in[13]; const float* bc2 = (const float*)d_in[14];
    const float* Wd3 = (const float*)d_in[15]; const float* bd3 = (const float*)d_in[16];
    const float* Wc3 = (const float*)d_in[17]; const float* bc3 = (const float*)d_in[18];
    const float* Wo  = (const float*)d_in[19]; const float* bo  = (const float*)d_in[20];

    float* out    = (float*)d_out;
    float* rec    = out;                 // 8*3*256*256 = 1572864
    float* routed = out + 1572864;       // 8*4*32*32   = 32768
    float* grain  = routed + 32768;      // 8*16*16     = 2048
    float* entout = grain + 2048;        // 8*16*16     = 2048

    char* ws = (char*)d_ws;
    size_t off = 0;
    auto take = [&](size_t bytes) -> void* {
        void* p = ws + off;
        off = (off + bytes + 255) & ~(size_t)255;
        return p;
    };
    float* thrp     = (float*)take(4);
    float* entws    = (float*)take(2048 * 4);
    float* finews   = (float*)take(32768 * 4);
    float* coarsews = (float*)take(8192 * 4);
    float* routedws = (float*)take(32768 * 4);
    unsigned short* pWd1 = (unsigned short*)take((size_t)256 * 256 * 9 * 2);
    unsigned short* pWc1 = (unsigned short*)take((size_t)256 * 256 * 9 * 2);
    unsigned short* pWd2 = (unsigned short*)take((size_t)128 * 256 * 9 * 2);
    unsigned short* pWc2 = (unsigned short*)take((size_t)128 * 128 * 9 * 2);
    unsigned short* pWd3 = (unsigned short*)take((size_t)128 * 128 * 9 * 2);
    unsigned short* pWc3 = (unsigned short*)take((size_t)128 * 128 * 9 * 2);
    // big ping-pong activation buffers (pair-interleaved bf16), worst-case sized
    size_t bigE = (size_t)8 * 128 * 256 * 256;          // elements (bf16)
    unsigned short* bufA = (unsigned short*)take(bigE * 2);
    unsigned short* bufZ = (unsigned short*)take(bigE * 2);
    unsigned short* bufD = (unsigned short*)take(bigE * 2);

    // ---- front end -------------------------------------------------------
    k_conv_enc<<<(32768 + 255) / 256, 256, 0, stream>>>(x, We0, be0, finews, 8, 32, 32);
    k_conv_enc<<<(8192  + 255) / 256, 256, 0, stream>>>(x, We1, be1, coarsews, 16, 16, 16);
    k_entropy<<<2048, 256, 0, stream>>>(x, entws, entout);
    k_median<<<1, 1024, 0, stream>>>(entws, thrp);
    k_route<<<128, 256, 0, stream>>>(finews, coarsews, entws, thrp, routed, routedws, grain);

    // ---- weight prepack --------------------------------------------------
    k_prepack<<<(256 * 256 * 9 + 255) / 256, 256, 0, stream>>>(Wd1, pWd1, 256, 256, 1);
    k_prepack<<<(256 * 256 * 9 + 255) / 256, 256, 0, stream>>>(Wc1, pWc1, 256, 256, 0);
    k_prepack<<<(128 * 256 * 9 + 255) / 256, 256, 0, stream>>>(Wd2, pWd2, 128, 256, 1);
    k_prepack<<<(128 * 128 * 9 + 255) / 256, 256, 0, stream>>>(Wc2, pWc2, 128, 128, 0);
    k_prepack<<<(128 * 128 * 9 + 255) / 256, 256, 0, stream>>>(Wd3, pWd3, 128, 128, 1);
    k_prepack<<<(128 * 128 * 9 + 255) / 256, 256, 0, stream>>>(Wc3, pWc3, 128, 128, 0);

    // ---- decoder ---------------------------------------------------------
    // Wi: routed[8,4,32,32] -> bufA (256ch, 32x32, bf16 pairs)
    k_conv_wi<<<(8 * 256 * 1024 + 255) / 256, 256, 0, stream>>>(routedws, Wi, bi, bufA);

    // Stage 1: 256ch 32->64  (Cin=256 -> CSLABS=8, W=64 -> LOGW=6)
    {
        size_t tz = (size_t)8 * 128 * 64 * 64;  // dwords
        k_zstuff<<<(unsigned)((tz + 255) / 256), 256, 0, stream>>>(
            (const unsigned int*)bufA, (unsigned int*)bufZ, 32, 32, tz);
        k_conv_wmma<8, 6><<<dim3(64 * 64 / 128, 256 / 32, 8), 256, 0, stream>>>(
            pWd1, (const unsigned int*)bufZ, bd1, bufD, 256, 1);
        k_conv_wmma<8, 6><<<dim3(64 * 64 / 128, 256 / 32, 8), 256, 0, stream>>>(
            pWc1, (const unsigned int*)bufD, bc1, bufA, 256, 1);
    }
    // Stage 2: 256ch 64 -> 128ch 128
    {
        size_t tz = (size_t)8 * 128 * 128 * 128;
        k_zstuff<<<(unsigned)((tz + 255) / 256), 256, 0, stream>>>(
            (const unsigned int*)bufA, (unsigned int*)bufZ, 64, 64, tz);
        k_conv_wmma<8, 7><<<dim3(128 * 128 / 128, 128 / 32, 8), 256, 0, stream>>>(
            pWd2, (const unsigned int*)bufZ, bd2, bufD, 128, 1);
        k_conv_wmma<4, 7><<<dim3(128 * 128 / 128, 128 / 32, 8), 256, 0, stream>>>(
            pWc2, (const unsigned int*)bufD, bc2, bufA, 128, 1);
    }
    // Stage 3: 128ch 128 -> 128ch 256
    {
        size_t tz = (size_t)8 * 64 * 256 * 256;
        k_zstuff<<<(unsigned)((tz + 255) / 256), 256, 0, stream>>>(
            (const unsigned int*)bufA, (unsigned int*)bufZ, 128, 128, tz);
        k_conv_wmma<4, 8><<<dim3(256 * 256 / 128, 128 / 32, 8), 256, 0, stream>>>(
            pWd3, (const unsigned int*)bufZ, bd3, bufD, 128, 1);
        k_conv_wmma<4, 8><<<dim3(256 * 256 / 128, 128 / 32, 8), 256, 0, stream>>>(
            pWc3, (const unsigned int*)bufD, bc3, bufA, 128, 1);
    }
    // Output head: tanh(conv(h, Wo, bo)) -> rec
    k_conv_wo<<<(8 * 3 * 65536 + 255) / 256, 256, 0, stream>>>(bufA, Wo, bo, rec);
}
